// ConvAttention1d_38079180046955
// MI455X (gfx1250) — compile-verified, matrix operand order checked by reading, not run on hardware
//
#include <hip/hip_runtime.h>
#include <hip/hip_bf16.h>
#include <math.h>

// ---- problem constants (from reference) ----
#define C_IN   128
#define C_OUT  256
#define LEN    4096
#define PATCH  64
#define PK     9
#define OKK    9
#define NP     64          // NUM_PATCHES
#define BATCH  32

// ---- tiling ----
#define TT      64         // output t columns per workgroup (aligned with patch size)
#define HALO_T  72         // sa columns needed: TT + OK-1
#define XT      80         // x columns staged: HALO_T + PK-1
#define XPITCH  81         // pad to odd -> conflict-free strided fp32 reads
#define BPITCH  136        // f16 pitch for sa tile; 272B rows -> conflict-free b128 reads
#define NCHUNK  4          // 128 ci / 32 per WMMA K

typedef __attribute__((ext_vector_type(16))) _Float16  v16h;
typedef __attribute__((ext_vector_type(8)))  float     v8f;
typedef __attribute__((ext_vector_type(4)))  unsigned int v4u;

union Frag16 {
    v16h h;
    v4u  u[2];
};

// ---------------------------------------------------------------------------
// Kernel 0: pack output_filters fp32 [C_OUT][C_IN][OK] -> f16 [OK][4][C_OUT][32]
// Inner 32 = ci within K-chunk, contiguous, so an A-fragment lane load is two
// 16B chunks at f16 offsets {0,16} (lanes 0-15) / {8,24} (lanes 16-31).
// ---------------------------------------------------------------------------
__global__ void ConvAttention1d_pack_weights(const float* __restrict__ of,
                                             _Float16* __restrict__ wf) {
    int idx  = blockIdx.x * blockDim.x + threadIdx.x;   // 9*4*256*32 = 294912
    int ci32 = idx & 31;
    int co   = (idx >> 5) & 255;
    int cc   = (idx >> 13) & 3;
    int k    = idx >> 15;
    if (k < OKK) {
        float v = of[((size_t)co * C_IN + cc * 32 + ci32) * OKK + k];
        wf[idx] = (_Float16)v;
    }
}

// ---------------------------------------------------------------------------
// Kernel 1: fused  (per-patch depthwise conv) -> f16 ->  (dense conv GEMM via
// WMMA, 9 shifted GEMMs accumulated in fp32) -> SELU.
// grid = (LEN/TT, BATCH), block = 256 (8 wave32)
// ---------------------------------------------------------------------------
__launch_bounds__(256)
__global__ void ConvAttention1d_38079180046955_kernel(
        const float*   __restrict__ x,    // [B][C_IN][LEN]
        const float*   __restrict__ pf,   // [C_IN][NP][1][PK]
        const _Float16* __restrict__ wf,  // packed weights [9][4][256][32]
        float*         __restrict__ out)  // [B][C_OUT][LEN]
{
    __shared__ float xs[C_IN * XPITCH];                      // 41472 B
    __shared__ __align__(16) _Float16 sb[HALO_T * BPITCH];   // 19584 B

    const int tid = threadIdx.x;
    const int t0  = blockIdx.x * TT;
    const int b   = blockIdx.y;
    const float* xb = x + (size_t)b * C_IN * LEN;

    // ---------- Phase A: stage x[ci][t0-8 .. t0+71] into LDS (fp32) ----------
    for (int idx = tid; idx < C_IN * XT; idx += 256) {
        int tt = idx % XT;
        int ci = idx / XT;
        int tg = t0 - 8 + tt;
        float v = 0.0f;
        if (tg >= 0 && tg < LEN) v = xb[(size_t)ci * LEN + tg];
        xs[ci * XPITCH + tt] = v;
    }
    __syncthreads();

    // ---------- Phase B: depthwise per-patch conv -> sa (f16) in LDS [t][ci] --
    // sa[tl][ci], tl = 0..71 corresponds to global t = t0 + tl - 4
    for (int it = 0; it < (HALO_T * C_IN) / 256; ++it) {
        int idx = it * 256 + tid;
        int ci  = idx & (C_IN - 1);
        int tl  = idx >> 7;                 // 0..71
        int tg  = t0 + tl - 4;
        float acc = 0.0f;
        if (tg >= 0 && tg < LEN) {
            int p = tg >> 6;                // patch index
            int u = tg & 63;                // position within patch
            const float* fp = pf + ((size_t)ci * NP + p) * PK;
            const float* xr = xs + ci * XPITCH + tl; // xs col (tl+kk) == x[tg+kk-4]
            #pragma unroll
            for (int kk = 0; kk < PK; ++kk) {
                int uu = u + kk - 4;
                if (uu >= 0 && uu < PATCH) acc += fp[kk] * xr[kk];
            }
        }
        sb[tl * BPITCH + ci] = (_Float16)acc;
    }
    __syncthreads();

    // ---------- Phase C: out[co,t] = sum_{k,ci} W[co,ci,k]*sa[ci,t+k-4] ------
    const int wave = tid >> 5;         // 0..7 -> co rows [wave*32, wave*32+32)
    const int lane = tid & 31;
    const int ln16 = lane & 15;
    const int hi8  = (lane >= 16) ? 8 : 0;

    const v8f vzero = {0.f, 0.f, 0.f, 0.f, 0.f, 0.f, 0.f, 0.f};
    v8f acc[2][4];
    #pragma unroll
    for (int i = 0; i < 2; ++i)
        #pragma unroll
        for (int j = 0; j < 4; ++j) acc[i][j] = vzero;

    for (int k = 0; k < OKK; ++k) {
        #pragma unroll
        for (int cc = 0; cc < NCHUNK; ++cc) {
            // A fragments: rows co = wave*32 + i*16 + ln16, K = ci chunk cc
            Frag16 a[2];
            #pragma unroll
            for (int i = 0; i < 2; ++i) {
                int co = wave * 32 + i * 16 + ln16;
                const _Float16* ap =
                    wf + ((((size_t)k * NCHUNK + cc) * C_OUT + co) << 5) + hi8;
                a[i].u[0] = *(const v4u*)(ap);
                a[i].u[1] = *(const v4u*)(ap + 16);
            }
            // B fragments: column n = j*16 + ln16, sa row tl = n + k (t+k-4 shift)
            Frag16 bf[4];
            #pragma unroll
            for (int j = 0; j < 4; ++j) {
                const _Float16* bp =
                    sb + (j * 16 + ln16 + k) * BPITCH + cc * 32 + hi8;
                bf[j].u[0] = *(const v4u*)(bp);
                bf[j].u[1] = *(const v4u*)(bp + 16);
            }
            #pragma unroll
            for (int i = 0; i < 2; ++i)
                #pragma unroll
                for (int j = 0; j < 4; ++j)
                    acc[i][j] = __builtin_amdgcn_wmma_f32_16x16x32_f16(
                        false, a[i].h, false, bf[j].h,
                        (short)0, acc[i][j], false, false);
        }
    }

    // ---------- Epilogue: SELU + store (C/D layout: VGPR e -> M = e or e+8) --
    const float kScale      = 1.0507009873554805f;
    const float kScaleAlpha = 1.7580993408473766f;   // scale * alpha
    float* ob = out + (size_t)b * C_OUT * LEN;
    #pragma unroll
    for (int i = 0; i < 2; ++i) {
        #pragma unroll
        for (int j = 0; j < 4; ++j) {
            int t = t0 + j * 16 + ln16;
            #pragma unroll
            for (int e = 0; e < 8; ++e) {
                int co = wave * 32 + i * 16 + e + hi8;
                float v = acc[i][j][e];
                v = (v > 0.0f) ? kScale * v : kScaleAlpha * (expf(v) - 1.0f);
                ob[(size_t)co * LEN + t] = v;
            }
        }
    }
}

extern "C" void kernel_launch(void* const* d_in, const int* in_sizes, int n_in,
                              void* d_out, int out_size, void* d_ws, size_t ws_size,
                              hipStream_t stream) {
    (void)in_sizes; (void)n_in; (void)out_size; (void)ws_size;
    const float* x  = (const float*)d_in[0];   // [32][128][4096]
    const float* pf = (const float*)d_in[1];   // [128][64][1][9]
    const float* of = (const float*)d_in[2];   // [256][128][9]
    float* out = (float*)d_out;                // [32][256][4096]
    _Float16* wf = (_Float16*)d_ws;            // 576 KB packed weights

    // Kernel 0: pack weights to f16 WMMA layout (9*4*256*32 elems / 256 = 1152 blocks)
    ConvAttention1d_pack_weights<<<1152, 256, 0, stream>>>(of, wf);

    // Kernel 1: fused depthwise + WMMA GEMM + SELU
    dim3 grid(LEN / TT, BATCH);
    ConvAttention1d_38079180046955_kernel<<<grid, 256, 0, stream>>>(x, pf, wf, out);
}
